// LSTMModel_9663676416622
// MI455X (gfx1250) — matrix operand order repaired and checked by reading, not compiled, not run on hardware
//
#include <hip/hip_runtime.h>
#include <hip/hip_bf16.h>
#include <math.h>

#define D_IN   512
#define HID    1024
#define GATES  (4 * HID)
#define BSZ    64
#define TLEN   512
#define BH     (BSZ * HID)

typedef __attribute__((ext_vector_type(16))) __bf16 v16bf;
typedef __attribute__((ext_vector_type(8)))  __bf16 v8bf;
typedef __attribute__((ext_vector_type(8)))  float  v8f;

union AFrag { v16bf v; v8bf h[2]; };

static __device__ __forceinline__ __bf16 f2bf(float x) {
  union { float f; unsigned u; } v; v.f = x;
  unsigned r = v.u + 0x7FFFu + ((v.u >> 16) & 1u);   // round-to-nearest-even
  unsigned short s = (unsigned short)(r >> 16);
  return __builtin_bit_cast(__bf16, s);
}

#if __has_builtin(__builtin_amdgcn_tanhf)
static __device__ __forceinline__ float fast_tanh(float x) {
  return __builtin_amdgcn_tanhf(x);    // v_tanh_f32 (gfx1250 TRANS op)
}
#else
static __device__ __forceinline__ float fast_tanh(float x) { return tanhf(x); }
#endif

static __device__ __forceinline__ float sigmoidf_(float x) {
  return 1.0f / (1.0f + __expf(-x));
}

// ---- one-time prep kernels -------------------------------------------------

__global__ void cvt_f32_to_bf16(const float* __restrict__ src,
                                __bf16* __restrict__ dst, int n) {
  int i = blockIdx.x * blockDim.x + threadIdx.x;
  int stride = gridDim.x * blockDim.x;
  for (; i < n; i += stride) dst[i] = f2bf(src[i]);
}

__global__ void bias_combine(const float* __restrict__ a,
                             const float* __restrict__ b,
                             float* __restrict__ dst, int n) {
  int i = blockIdx.x * blockDim.x + threadIdx.x;
  if (i < n) dst[i] = a[i] + b[i];
}

// ---- persistent per-layer LSTM kernel -------------------------------------
// Grid: 64 blocks (= hidden-unit tiles) x 256 threads (8 waves).
//   waves 0..3 : batch tile w,   K-half 0
//   waves 4..7 : batch tile w-4, K-half 1  (partials reduced through LDS)
// The block's 64 W rows (4 gates x 16 units, full K = K_IN+HID, w_ih|w_hh
// concatenated) are staged into LDS once; B fragments then come from
// ds_load_b128 with immediate offsets. One device-wide barrier per step;
// h ping-pongs, c is owner-exclusive.

template <int K_IN, bool WRITE_ALL, bool GATHER>
__global__ void __launch_bounds__(256)
lstm_layer(const __bf16* __restrict__ ain, int aTimeStride, int aBatchStride,
           const __bf16* __restrict__ wih,   // [4H, K_IN] row-major bf16
           const __bf16* __restrict__ whh,   // [4H, HID]  row-major bf16
           const float*  __restrict__ bias,  // [4H] combined
           __bf16* __restrict__ hping,       // [2, B, HID] bf16 (zeroed)
           float*  __restrict__ cst,         // [B, HID] f32 (zeroed)
           __bf16* __restrict__ hall,        // [T, B, HID] bf16 out (layer 0)
           const int* __restrict__ seqlen,   // [B] (layer 1)
           float* __restrict__ out,          // [B, HID] f32 (layer 1)
           unsigned* __restrict__ barcnt) {
  constexpr int KTOT  = K_IN + HID;
  constexpr int KHALF = KTOT / 2;
  constexpr size_t WELEMS = (size_t)64 * KTOT;           // LDS W slab elements

  extern __shared__ char smem[];
  __bf16* lw   = (__bf16*)smem;                           // [64][KTOT]
  float*  pacc = (float*)(smem + WELEMS * 2);             // [4][4][32][8] f32

  const int lane  = threadIdx.x & 31;
  const int wv    = threadIdx.x >> 5;      // 0..7
  const int mtile = wv & 3;
  const int khalf = wv >> 2;
  const int mbase = mtile * 16;
  const int ubase = blockIdx.x * 16;       // hidden-unit tile base
  const int l16   = lane & 15;
  const int lhi   = lane >> 4;
  const int aoff  = lhi * 8;               // A-fragment half-lane K offset
  const unsigned nblk = gridDim.x;

  // ---- stage this block's W rows into LDS (once) ----
  for (int idx = threadIdx.x; idx < 64 * (KTOT / 8); idx += blockDim.x) {
    const int row = idx / (KTOT / 8);
    const int col = (idx % (KTOT / 8)) * 8;
    const int g = row >> 4, n16 = row & 15;
    const __bf16* src =
        (col < K_IN)
            ? (wih + (size_t)(g * HID + ubase + n16) * K_IN + col)
            : (whh + (size_t)(g * HID + ubase + n16) * HID + (col - K_IN));
    *(v8bf*)(lw + (size_t)row * KTOT + col) = *(const v8bf*)src;
  }
  __syncthreads();

  // Per-lane LDS base per gate: row = g*16 + l16 (N=lane%16), +lhi*8 K bias.
  const __bf16* lbg[4];
#pragma unroll
  for (int g = 0; g < 4; ++g)
    lbg[g] = lw + (size_t)(g * 16 + l16) * KTOT + lhi * 8;

  float bs[4];
#pragma unroll
  for (int g = 0; g < 4; ++g) bs[g] = bias[g * HID + ubase + l16];

  int lens[8];
  if (GATHER && khalf == 0) {
#pragma unroll
    for (int r = 0; r < 8; ++r) lens[r] = seqlen[mbase + r + 8 * lhi];
  }

  // 32-bit per-lane A element offsets (uniform base + voffset addressing).
  const int axOff = (mbase + l16) * aBatchStride + aoff;
  const int ahOff = (mbase + l16) * HID + aoff;

  // This wave-group's K range, split at the x|h boundary K_IN.
  const int kbeg = khalf * KHALF;
  const int kend = kbeg + KHALF;
  const int kx_end = (kend < K_IN) ? kend : K_IN;  // x part: [kbeg, kx_end)
  const int kh_beg = (kbeg > K_IN) ? kbeg : K_IN;  // h part: [kh_beg, kend)

  const v8f vzero = {0.f, 0.f, 0.f, 0.f, 0.f, 0.f, 0.f, 0.f};

  for (int t = 0; t < TLEN; ++t) {
    const __bf16* hprev = hping + (size_t)(t & 1) * BH;
    __bf16*       hnext = hping + (size_t)((t + 1) & 1) * BH;
    const __bf16* abase = ain + (size_t)t * aTimeStride;

    v8f acc[4];
#pragma unroll
    for (int g = 0; g < 4; ++g) acc[g] = vzero;

    // gates += x_t @ W_ih^T  (A from global, B from LDS)
#pragma unroll 4
    for (int k = kbeg; k < kx_end; k += 32) {
      AFrag a;
      a.h[0] = *(const v8bf*)(abase + axOff + k);
      a.h[1] = *(const v8bf*)(abase + axOff + k + 16);
#pragma unroll
      for (int g = 0; g < 4; ++g) {
        AFrag b;
        b.h[0] = *(const v8bf*)(lbg[g] + k);
        b.h[1] = *(const v8bf*)(lbg[g] + k + 16);
        acc[g] = __builtin_amdgcn_wmma_f32_16x16x32_bf16(
            false, a.v, false, b.v, (short)0, acc[g], false, false);
      }
    }

    // gates += h_{t-1} @ W_hh^T
#pragma unroll 4
    for (int k = kh_beg; k < kend; k += 32) {
      AFrag a;
      a.h[0] = *(const v8bf*)(hprev + ahOff + (k - K_IN));
      a.h[1] = *(const v8bf*)(hprev + ahOff + (k - K_IN) + 16);
#pragma unroll
      for (int g = 0; g < 4; ++g) {
        AFrag b;
        b.h[0] = *(const v8bf*)(lbg[g] + k);
        b.h[1] = *(const v8bf*)(lbg[g] + k + 16);
        acc[g] = __builtin_amdgcn_wmma_f32_16x16x32_bf16(
            false, a.v, false, b.v, (short)0, acc[g], false, false);
      }
    }

    // K-split reduction through LDS: waves 4..7 publish partials.
    if (khalf == 1) {
#pragma unroll
      for (int g = 0; g < 4; ++g)
        *(v8f*)(pacc + (size_t)(((mtile * 4 + g) * 32) + lane) * 8) = acc[g];
    }
    __syncthreads();

    if (khalf == 0) {
#pragma unroll
      for (int g = 0; g < 4; ++g)
        acc[g] += *(const v8f*)(pacc + (size_t)(((mtile * 4 + g) * 32) + lane) * 8);

      // Cell update. C/D layout: vgpr r -> M = r + 8*(lane/16), N = lane%16.
#pragma unroll
      for (int r = 0; r < 8; ++r) {
        const int b = mbase + r + 8 * lhi;
        const size_t idx = (size_t)b * HID + ubase + l16;
        const float ig = sigmoidf_(acc[0][r] + bs[0]);
        const float fg = sigmoidf_(acc[1][r] + bs[1]);
        const float gg = fast_tanh(acc[2][r] + bs[2]);
        const float og = sigmoidf_(acc[3][r] + bs[3]);
        const float cn = fg * cst[idx] + ig * gg;
        cst[idx] = cn;
        const float hn = og * fast_tanh(cn);
        hnext[idx] = f2bf(hn);
        if (WRITE_ALL) hall[(size_t)t * BH + idx] = f2bf(hn);
        if (GATHER) {
          if (t == lens[r] - 1) out[idx] = hn;
        }
      }
    }

    // Device-wide barrier: all writes of step t visible before step t+1 reads.
    if (t != TLEN - 1) {
      __threadfence();
      __syncthreads();
      if (threadIdx.x == 0) {
        atomicAdd(barcnt, 1u);
        const unsigned target = (unsigned)(t + 1) * nblk;
        while (*((volatile unsigned*)barcnt) < target) {
        }
      }
      __syncthreads();
      __threadfence();
    }
  }
}

// ---- host launch -----------------------------------------------------------

extern "C" void kernel_launch(void* const* d_in, const int* in_sizes, int n_in,
                              void* d_out, int out_size, void* d_ws,
                              size_t ws_size, hipStream_t stream) {
  const float* x      = (const float*)d_in[0];
  const float* w_ih_0 = (const float*)d_in[1];
  const float* w_hh_0 = (const float*)d_in[2];
  const float* b_ih_0 = (const float*)d_in[3];
  const float* b_hh_0 = (const float*)d_in[4];
  const float* w_ih_1 = (const float*)d_in[5];
  const float* w_hh_1 = (const float*)d_in[6];
  const float* b_ih_1 = (const float*)d_in[7];
  const float* b_hh_1 = (const float*)d_in[8];
  const int*   seqlen = (const int*)d_in[9];
  float* out = (float*)d_out;

  // Workspace carve (256-byte aligned slices).
  char* p = (char*)d_ws;
  auto carve = [&](size_t bytes) {
    char* r = p;
    p += (bytes + 255) & ~(size_t)255;
    return r;
  };
  __bf16* xbf    = (__bf16*)carve((size_t)BSZ * TLEN * D_IN * 2);
  __bf16* wih0   = (__bf16*)carve((size_t)GATES * D_IN * 2);
  __bf16* whh0   = (__bf16*)carve((size_t)GATES * HID * 2);
  __bf16* wih1   = (__bf16*)carve((size_t)GATES * HID * 2);
  __bf16* whh1   = (__bf16*)carve((size_t)GATES * HID * 2);
  float*  bias0  = (float*)carve((size_t)GATES * 4);
  float*  bias1  = (float*)carve((size_t)GATES * 4);
  __bf16* hall   = (__bf16*)carve((size_t)TLEN * BH * 2);
  __bf16* hping0 = (__bf16*)carve((size_t)2 * BH * 2);
  __bf16* hping1 = (__bf16*)carve((size_t)2 * BH * 2);
  float*  c0     = (float*)carve((size_t)BH * 4);
  float*  c1     = (float*)carve((size_t)BH * 4);
  unsigned* bar  = (unsigned*)carve(512);

  // Dynamic-LDS sizes: W slab + 16KB partial-accumulator region.
  const int smem0 = 64 * (D_IN + HID) * 2 + 4 * 4 * 32 * 8 * 4;   // 212992
  const int smem1 = 64 * (HID + HID) * 2 + 4 * 4 * 32 * 8 * 4;    // 278528
  hipFuncSetAttribute((const void*)lstm_layer<D_IN, true, false>,
                      hipFuncAttributeMaxDynamicSharedMemorySize, smem0);
  hipFuncSetAttribute((const void*)lstm_layer<HID, false, true>,
                      hipFuncAttributeMaxDynamicSharedMemorySize, smem1);

  // fp32 -> bf16 conversion (one-time HBM pass; then everything lives in L2).
  {
    const int thr = 256;
    int n;
    n = BSZ * TLEN * D_IN;
    cvt_f32_to_bf16<<<2048, thr, 0, stream>>>(x, xbf, n);
    n = GATES * D_IN;
    cvt_f32_to_bf16<<<(n + thr - 1) / thr, thr, 0, stream>>>(w_ih_0, wih0, n);
    n = GATES * HID;
    cvt_f32_to_bf16<<<(n + thr - 1) / thr, thr, 0, stream>>>(w_hh_0, whh0, n);
    cvt_f32_to_bf16<<<(n + thr - 1) / thr, thr, 0, stream>>>(w_ih_1, wih1, n);
    cvt_f32_to_bf16<<<(n + thr - 1) / thr, thr, 0, stream>>>(w_hh_1, whh1, n);
    bias_combine<<<(GATES + thr - 1) / thr, thr, 0, stream>>>(b_ih_0, b_hh_0,
                                                              bias0, GATES);
    bias_combine<<<(GATES + thr - 1) / thr, thr, 0, stream>>>(b_ih_1, b_hh_1,
                                                              bias1, GATES);
  }

  // Zero states and barrier counters.
  hipMemsetAsync(hping0, 0, (size_t)2 * BH * 2, stream);
  hipMemsetAsync(hping1, 0, (size_t)2 * BH * 2, stream);
  hipMemsetAsync(c0, 0, (size_t)BH * 4, stream);
  hipMemsetAsync(c1, 0, (size_t)BH * 4, stream);
  hipMemsetAsync(bar, 0, 512, stream);

  // Layer 0: x [B,T,D] bf16 -> row(b,t) = xbf + b*(T*D) + t*D
  lstm_layer<D_IN, true, false><<<64, 256, smem0, stream>>>(
      xbf, D_IN, TLEN * D_IN, wih0, whh0, bias0, hping0, c0, hall,
      nullptr, nullptr, bar);

  // Layer 1: h0 history [T,B,H] -> row(b,t) = hall + t*B*H + b*H
  lstm_layer<HID, false, true><<<64, 256, smem1, stream>>>(
      hall, BH, HID, wih1, whh1, bias1, hping1, c1, nullptr,
      seqlen, out, bar + 64);

  (void)in_sizes; (void)n_in; (void)out_size; (void)ws_size;
}
